// MultiHeadAttention_34265249087539
// MI455X (gfx1250) — matrix operand order repaired and checked by reading, not compiled, not run on hardware
//
#include <hip/hip_runtime.h>
#include <hip/hip_bf16.h>

typedef __bf16 bf16;
typedef __attribute__((ext_vector_type(16))) __bf16 v16bf;
typedef __attribute__((ext_vector_type(8)))  __bf16 v8bf;
typedef __attribute__((ext_vector_type(8)))  float   v8f;
typedef unsigned int u32x4 __attribute__((ext_vector_type(4)));
typedef int          i32x8 __attribute__((ext_vector_type(8)));
typedef int          i32x4 __attribute__((ext_vector_type(4)));

#if defined(__has_include)
#if __has_include(<hip/amd_detail/amd_gfx1250_TDM.h>)
#define TDM_ARITY6 1
#endif
#endif

constexpr int Bc = 4;
constexpr int Sc = 2048;
constexpr int Dc = 1024;
constexpr int Hc = 16;
constexpr int HDc = 64;
constexpr float NEG_BIG = -1e30f;

// ---------------------------------------------------------------------------
// Fragment loaders (16-bit WMMA, wave32; A/B in this code's contiguous k order)
// A (16x32, MxK): lane l -> row l&15 ; K = e + 16*(l>>4)
// B (32x16, KxN): lane l -> col l&15 ; K = e + 16*(l>>4)  (K-major source)
// C/D f32 16x16:  lane l -> col l&15 ; vgpr r -> row r + 8*(l>>4)
// ---------------------------------------------------------------------------
__device__ __forceinline__ v16bf load_fragA(const bf16* __restrict__ p, int row0,
                                            int ld, int kbase, int lane) {
  int m  = row0 + (lane & 15);
  int hi = lane >> 4;
  const bf16* q = p + (size_t)m * ld + kbase + 16 * hi;
  v8bf lo = *(const v8bf*)q;
  v8bf hg = *(const v8bf*)(q + 8);
  v16bf out;
#pragma unroll
  for (int i = 0; i < 8; ++i) { out[i] = lo[i]; out[i + 8] = hg[i]; }
  return out;
}

__device__ __forceinline__ v16bf load_fragB(const bf16* __restrict__ p, int col0,
                                            int ld, int kbase, int lane) {
  int n  = col0 + (lane & 15);
  int hi = lane >> 4;
  const bf16* q = p + (size_t)n * ld + kbase + 16 * hi;
  v8bf lo = *(const v8bf*)q;
  v8bf hg = *(const v8bf*)(q + 8);
  v16bf out;
#pragma unroll
  for (int i = 0; i < 8; ++i) { out[i] = lo[i]; out[i + 8] = hg[i]; }
  return out;
}

__device__ __forceinline__ v8f wmma_bf16(v16bf a, v16bf b, v8f c) {
  return __builtin_amdgcn_wmma_f32_16x16x32_bf16(false, a, false, b,
                                                 (short)0, c, false, false);
}

// ---------------------------------------------------------------------------
// Tensor Data Mover: 2D tile (tile_w x tile_h elements, 2B each) from global
// (row stride = stride_elems) into LDS at byte offset lds_off. D# per ISA §8.
// ---------------------------------------------------------------------------
__device__ __forceinline__ unsigned int lds_off_of(const void* p) {
  return (unsigned int)(uintptr_t)p;   // LDS aperture: low 32 bits = LDS offset
}

__device__ __forceinline__ void tdm_load_2d(unsigned int lds_off,
                                            const void* gaddr,
                                            unsigned int tile_w,
                                            unsigned int tile_h,
                                            unsigned int stride_elems) {
  unsigned long long ga = (unsigned long long)(uintptr_t)gaddr;
  u32x4 g0;
  g0[0] = 1u;                                        // count=1 (valid), user mode
  g0[1] = lds_off;                                   // lds_addr (bytes)
  g0[2] = (unsigned int)ga;                          // global_addr[31:0]
  g0[3] = (unsigned int)((ga >> 32) & 0x01FFFFFFu)   // global_addr[56:32]
          | (2u << 30);                              // type = 2 ("image")
  i32x8 g1;
  g1[0] = 0x00010000;                 // workgroup_mask=0, data_size=1 (2 bytes)
  g1[1] = (int)(tile_w << 16);        // tensor_dim0[15:0] in [31:16]
  g1[2] = (int)(tile_h << 16);        // tensor_dim1[15:0] in [31:16]
  g1[3] = (int)(tile_w << 16);        // tile_dim0 in [31:16]
  g1[4] = (int)tile_h;                // tile_dim1 in [15:0], tile_dim2=0
  g1[5] = (int)stride_elems;          // tensor_dim0_stride[31:0]
  g1[6] = 0;
  g1[7] = 0;
  i32x4 z4 = {};
#if defined(TDM_ARITY6)
  i32x8 z8 = {};
  __builtin_amdgcn_tensor_load_to_lds(g0, g1, z4, z4, z8, 0);
#else
  __builtin_amdgcn_tensor_load_to_lds(g0, g1, z4, z4, 0);
#endif
}

// ---------------------------------------------------------------------------
// fp32 -> bf16 convert
// ---------------------------------------------------------------------------
__global__ void k_cvt_bf16(const float* __restrict__ src, bf16* __restrict__ dst,
                           int n) {
  for (int i = blockIdx.x * blockDim.x + threadIdx.x; i < n;
       i += gridDim.x * blockDim.x)
    dst[i] = (bf16)src[i];
}

// W [D_in, D_out] fp32 -> Wt [D_out, D_in] bf16 (K-contiguous per output col)
__global__ void k_transpose_w(const float* __restrict__ w, bf16* __restrict__ wt,
                              int d) {
  int i = blockIdx.x * blockDim.x + threadIdx.x;
  if (i >= d * d) return;
  int o  = i / d;
  int in = i - o * d;
  wt[(size_t)o * d + in] = (bf16)w[(size_t)in * d + o];
}

// ---------------------------------------------------------------------------
// Wave-tiled GEMM: C[M,N] = A[M,K] * Bt[N,K]^T  (bf16 in, f32 acc)
// Each wave owns a 32x64 tile: 2 A-fragments x 4 B-fragments = 8 WMMA per
// 32-wide K step. True ping-pong pipeline: two named register sets consumed
// then refilled in place (no copy rotation -> no mov chains, no spills).
// Requires K % 64 == 0 and K >= 128 (K = 1024 here).
// mode 0: bf16 out row-major [M,N]
// mode 1: bf16 out scattered as Vt[b][h][hd][S]   (V projection)
// mode 2: f32 out row-major [M,N] + bias          (output projection)
// ---------------------------------------------------------------------------
__global__ void __launch_bounds__(256, 1)
k_gemm(const bf16* __restrict__ A, const bf16* __restrict__ Bt,
       int M, int N, int K, bf16* __restrict__ outb, int mode,
       float* __restrict__ outf, const float* __restrict__ bias) {
  int lane = threadIdx.x & 31;
  int wave = blockIdx.x * (blockDim.x >> 5) + (threadIdx.x >> 5);
  int tilesN = N >> 6;
  int tm = wave / tilesN;
  int tn = wave - tm * tilesN;
  int m0 = tm * 32, n0 = tn * 64;
  if (m0 >= M) return;

  v8f acc[2][4] = {};

  // ping-pong register sets: set X covers k, set Y covers k+32
  v16bf a0X = load_fragA(A, m0,      K, 0,  lane);
  v16bf a1X = load_fragA(A, m0 + 16, K, 0,  lane);
  v16bf a0Y = load_fragA(A, m0,      K, 32, lane);
  v16bf a1Y = load_fragA(A, m0 + 16, K, 32, lane);
  v16bf bX[4], bY[4];
#pragma unroll
  for (int c = 0; c < 4; ++c) {
    bX[c] = load_fragB(Bt, n0 + 16 * c, K, 0,  lane);
    bY[c] = load_fragB(Bt, n0 + 16 * c, K, 32, lane);
  }

  for (int k = 0; k < K - 64; k += 64) {
    // far-ahead prefetch into L2 (global_prefetch_b8)
    if (k + 256 < K) {
      __builtin_prefetch(A  + (size_t)(m0 + (lane & 15)) * K + k + 256, 0, 1);
      __builtin_prefetch(Bt + (size_t)(n0 + (lane & 15)) * K + k + 256, 0, 1);
    }
    // consume set X (k), refill with k+64
#pragma unroll
    for (int c = 0; c < 4; ++c) {
      acc[0][c] = wmma_bf16(a0X, bX[c], acc[0][c]);
      acc[1][c] = wmma_bf16(a1X, bX[c], acc[1][c]);
    }
    a0X = load_fragA(A, m0,      K, k + 64, lane);
    a1X = load_fragA(A, m0 + 16, K, k + 64, lane);
#pragma unroll
    for (int c = 0; c < 4; ++c)
      bX[c] = load_fragB(Bt, n0 + 16 * c, K, k + 64, lane);

    // consume set Y (k+32), refill with k+96
#pragma unroll
    for (int c = 0; c < 4; ++c) {
      acc[0][c] = wmma_bf16(a0Y, bY[c], acc[0][c]);
      acc[1][c] = wmma_bf16(a1Y, bY[c], acc[1][c]);
    }
    a0Y = load_fragA(A, m0,      K, k + 96, lane);
    a1Y = load_fragA(A, m0 + 16, K, k + 96, lane);
#pragma unroll
    for (int c = 0; c < 4; ++c)
      bY[c] = load_fragB(Bt, n0 + 16 * c, K, k + 96, lane);
  }
  // epilogue: k = K-64 (set X) and k = K-32 (set Y)
#pragma unroll
  for (int c = 0; c < 4; ++c) {
    acc[0][c] = wmma_bf16(a0X, bX[c], acc[0][c]);
    acc[1][c] = wmma_bf16(a1X, bX[c], acc[1][c]);
  }
#pragma unroll
  for (int c = 0; c < 4; ++c) {
    acc[0][c] = wmma_bf16(a0Y, bY[c], acc[0][c]);
    acc[1][c] = wmma_bf16(a1Y, bY[c], acc[1][c]);
  }

  int hi = lane >> 4, ln = lane & 15;
#pragma unroll
  for (int g = 0; g < 2; ++g) {
#pragma unroll
    for (int c = 0; c < 4; ++c) {
#pragma unroll
      for (int r = 0; r < 8; ++r) {
        int Mi = m0 + 16 * g + r + 8 * hi;
        int Ni = n0 + 16 * c + ln;
        float v = acc[g][c][r];
        if (mode == 0) {
          outb[(size_t)Mi * N + Ni] = (bf16)v;
        } else if (mode == 1) {
          int b_ = Mi / Sc, s_ = Mi - b_ * Sc;
          int h_ = Ni >> 6, d_ = Ni & 63;
          outb[(((size_t)b_ * Hc + h_) * HDc + d_) * Sc + s_] = (bf16)v;
        } else {
          outf[(size_t)Mi * N + Ni] = v + bias[Ni];
        }
      }
    }
  }
}

// ---------------------------------------------------------------------------
// Flash attention (causal), one wave per (b, h, 16-query tile).
// S^T = K*Q^T so softmax stats live in lane columns; ctx^T = V^T*P^T with P^T
// packed via one paired-lane exchange. K and V^T 32-kv blocks are staged into
// wave-private LDS by the Tensor Data Mover, double-buffered, synchronized
// with s_wait_tensorcnt (no barriers needed: regions are wave-private).
// ---------------------------------------------------------------------------
__global__ void __launch_bounds__(128, 1)
k_attn(const bf16* __restrict__ Qb, const bf16* __restrict__ Kb,
       const bf16* __restrict__ Vt, bf16* __restrict__ ctxb) {
  // per wave: K tiles 2x(32x64) + V tiles 2x(64x32) = 8192 bf16 = 16 KiB
  __shared__ __align__(16) bf16 smem[4 * 8192];   // 4 waves * 16 KiB = 64 KiB

  int lane = threadIdx.x & 31;
  int wslot = threadIdx.x >> 5;
  int wave = blockIdx.x * (blockDim.x >> 5) + wslot;
  int qt = wave & (Sc / 16 - 1);          // 128 q-tiles
  int h  = (wave >> 7) & (Hc - 1);
  int b  = wave >> 11;
  if (b >= Bc) return;

  bf16* kbuf0 = smem + wslot * 8192;
  bf16* kbuf1 = kbuf0 + 2048;
  bf16* vbuf0 = kbuf0 + 4096;
  bf16* vbuf1 = kbuf0 + 6144;

  int q0 = qt << 4;
  int hi = lane >> 4, ln = lane & 15;
  int q  = q0 + ln;

  const bf16* qbase = Qb + (size_t)b * Sc * Dc + h * HDc;
  const bf16* kbase = Kb + (size_t)b * Sc * Dc + h * HDc;
  const bf16* vbase = Vt + ((size_t)b * Hc + h) * HDc * Sc;

  // Q^T B-fragments for the two 32-wide d chunks (hd = 64)
  v16bf qf0 = load_fragB(qbase, q0, Dc, 0, lane);
  v16bf qf1 = load_fragB(qbase, q0, Dc, 32, lane);

  v8f ctx[4] = {};
  float mrun = NEG_BIG, lsum = 0.f;

  int nblk = (q0 + 16 + 31) >> 5;         // cover kv <= q0+15

  // TDM: stage block 0 (K: 32 rows x 64 d, stride D; V^T: 64 rows x 32 kv, stride S)
  tdm_load_2d(lds_off_of(kbuf0), kbase, 64, 32, Dc);
  tdm_load_2d(lds_off_of(vbuf0), vbase, 32, 64, Sc);

  for (int blk = 0; blk < nblk; ++blk) {
    int k0 = blk << 5;
    bf16* kb_l = (blk & 1) ? kbuf1 : kbuf0;
    bf16* vb_l = (blk & 1) ? vbuf1 : vbuf0;

    if (blk + 1 < nblk) {
      int kn = (blk + 1) << 5;
      bf16* kb_n = ((blk + 1) & 1) ? kbuf1 : kbuf0;
      bf16* vb_n = ((blk + 1) & 1) ? vbuf1 : vbuf0;
      tdm_load_2d(lds_off_of(kb_n), kbase + (size_t)kn * Dc, 64, 32, Dc);
      tdm_load_2d(lds_off_of(vb_n), vbase + kn, 32, 64, Sc);
      __builtin_amdgcn_s_wait_tensorcnt(2);   // current block's 2 ops done
    } else {
      __builtin_amdgcn_s_wait_tensorcnt(0);
    }
    asm volatile("" ::: "memory");

    // S^T tiles from LDS K-tile (ld=64): c0 -> kv [k0,k0+16), c1 -> +16
    v8f c0 = {}, c1 = {};
    {
      v16bf ka = load_fragA(kb_l, 0, 64, 0, lane);
      c0 = wmma_bf16(ka, qf0, c0);
      ka = load_fragA(kb_l, 0, 64, 32, lane);
      c0 = wmma_bf16(ka, qf1, c0);
      ka = load_fragA(kb_l, 16, 64, 0, lane);
      c1 = wmma_bf16(ka, qf0, c1);
      ka = load_fragA(kb_l, 16, 64, 32, lane);
      c1 = wmma_bf16(ka, qf1, c1);
    }

    // scale + causal mask
    bool partial = (k0 + 31) > q0;
    float s0[8], s1[8];
#pragma unroll
    for (int r = 0; r < 8; ++r) {
      int kv0 = k0 + r + 8 * hi;
      int kv1 = kv0 + 16;
      float x0 = c0[r] * 0.125f;          // 1/sqrt(64)
      float x1 = c1[r] * 0.125f;
      if (partial) {
        if (kv0 > q) x0 = NEG_BIG;
        if (kv1 > q) x1 = NEG_BIG;
      }
      s0[r] = x0; s1[r] = x1;
    }

    // online softmax (per-q column split across lane pair l, l^16)
    float mloc = NEG_BIG;
#pragma unroll
    for (int r = 0; r < 8; ++r) mloc = fmaxf(mloc, fmaxf(s0[r], s1[r]));
    mloc = fmaxf(mloc, __shfl_xor(mloc, 16, 32));
    float mnew  = fmaxf(mrun, mloc);
    float alpha = __expf(mrun - mnew);

    float p0[8], p1[8], psum = 0.f;
#pragma unroll
    for (int r = 0; r < 8; ++r) {
      p0[r] = __expf(s0[r] - mnew);
      p1[r] = __expf(s1[r] - mnew);
      psum += p0[r] + p1[r];
    }
    psum += __shfl_xor(psum, 16, 32);
    lsum = lsum * alpha + psum;
    mrun = mnew;

#pragma unroll
    for (int g = 0; g < 4; ++g)
#pragma unroll
      for (int r = 0; r < 8; ++r) ctx[g][r] *= alpha;

    // Pack P^T B-fragment: B wants kv = e + 16*hi; score frags hold
    // kv = {r+8hi, 16+r+8hi} -> exchange halves with the paired lane.
    v16bf pb;
#pragma unroll
    for (int r = 0; r < 8; ++r) {
      float o0 = __shfl_xor(p0[r], 16, 32);
      float o1 = __shfl_xor(p1[r], 16, 32);
      pb[r]     = (bf16)(hi ? o1 : p0[r]);
      pb[r + 8] = (bf16)(hi ? p1[r] : o0);
    }

    // ctx^T += V^T * P^T from LDS V-tile (ld=32), 4 hd-row groups of 16
#pragma unroll
    for (int g = 0; g < 4; ++g) {
      v16bf va = load_fragA(vb_l, g * 16, 32, 0, lane);
      ctx[g] = wmma_bf16(va, pb, ctx[g]);
    }
  }

  // normalize + store ctx in [B,S,D] bf16 (16B per lane per group)
  float inv = 1.0f / lsum;
#pragma unroll
  for (int g = 0; g < 4; ++g) {
    v8bf o;
#pragma unroll
    for (int r = 0; r < 8; ++r) o[r] = (bf16)(ctx[g][r] * inv);
    bf16* dst = ctxb + ((size_t)b * Sc + q) * Dc + h * HDc + g * 16 + 8 * hi;
    *(v8bf*)dst = o;
  }
}

// ---------------------------------------------------------------------------
extern "C" void kernel_launch(void* const* d_in, const int* in_sizes, int n_in,
                              void* d_out, int out_size, void* d_ws,
                              size_t ws_size, hipStream_t stream) {
  (void)in_sizes; (void)n_in; (void)out_size; (void)ws_size;
  const float* x  = (const float*)d_in[0];
  const float* Wq = (const float*)d_in[1];
  const float* Wk = (const float*)d_in[2];
  const float* Wv = (const float*)d_in[3];
  const float* Wo = (const float*)d_in[4];
  const float* bo = (const float*)d_in[5];
  float* out = (float*)d_out;

  constexpr size_t MB = 1ull << 20;
  char* ws = (char*)d_ws;
  bf16* xb   = (bf16*)(ws + 0 * MB);    // 16 MiB  x in bf16
  bf16* wqt  = (bf16*)(ws + 16 * MB);   //  2 MiB  Wq^T
  bf16* wkt  = (bf16*)(ws + 18 * MB);
  bf16* wvt  = (bf16*)(ws + 20 * MB);
  bf16* wot  = (bf16*)(ws + 22 * MB);
  bf16* qb   = (bf16*)(ws + 24 * MB);   // 16 MiB  Q [B,S,D]
  bf16* kb   = (bf16*)(ws + 40 * MB);   // 16 MiB  K [B,S,D]
  bf16* vt   = (bf16*)(ws + 56 * MB);   // 16 MiB  V^T [B,H,hd,S]
  bf16* ctxb = (bf16*)(ws + 72 * MB);   // 16 MiB  ctx [B,S,D]

  const int M = Bc * Sc;                // 8192
  const int nElem = M * Dc;             // 8,388,608

  k_cvt_bf16<<<4096, 256, 0, stream>>>(x, xb, nElem);

  int tBlocks = (Dc * Dc + 255) / 256;
  k_transpose_w<<<tBlocks, 256, 0, stream>>>(Wq, wqt, Dc);
  k_transpose_w<<<tBlocks, 256, 0, stream>>>(Wk, wkt, Dc);
  k_transpose_w<<<tBlocks, 256, 0, stream>>>(Wv, wvt, Dc);
  k_transpose_w<<<tBlocks, 256, 0, stream>>>(Wo, wot, Dc);

  int gemmWaves  = (M / 32) * (Dc / 64);            // 4096 waves (32x64 tiles)
  int gemmBlocks = (gemmWaves + 7) / 8;             // 8 waves / 256-thr block
  k_gemm<<<gemmBlocks, 256, 0, stream>>>(xb, wqt, M, Dc, Dc, qb, 0, nullptr, nullptr);
  k_gemm<<<gemmBlocks, 256, 0, stream>>>(xb, wkt, M, Dc, Dc, kb, 0, nullptr, nullptr);
  k_gemm<<<gemmBlocks, 256, 0, stream>>>(xb, wvt, M, Dc, Dc, vt, 1, nullptr, nullptr);

  int attnWaves  = Bc * Hc * (Sc / 16);             // 8192 waves
  int attnBlocks = (attnWaves + 3) / 4;             // 4 waves / 128-thr block
  k_attn<<<attnBlocks, 128, 0, stream>>>(qb, kb, vt, ctxb);

  k_gemm<<<gemmBlocks, 256, 0, stream>>>(ctxb, wot, M, Dc, Dc, nullptr, 2, out, bo);
}
